// CornerNetModule_47270410060154
// MI455X (gfx1250) — compile-verified, matrix operand order checked by reading, not run on hardware
//
#include <hip/hip_runtime.h>
#include <hip/hip_bf16.h>

typedef __attribute__((ext_vector_type(16))) _Float16 v16h;
typedef __attribute__((ext_vector_type(8)))  float    v8f;

#define DEVINL __device__ __forceinline__

DEVINL v8f wmma_f16(v16h a, v16h b, v8f c) {
    return __builtin_amdgcn_wmma_f32_16x16x32_f16(false, a, false, b, (short)0, c, false, false);
}

// Load one lane's 16-half fragment from a k-contiguous source: per ISA layout the
// lane's values are rp[0..7] (K = k0+8*half .. +7) and rp[16..23] (K = k0+16+8*half .. +7).
// Caller passes rp = base + row*ld + k0 + 8*half. Both chunks are 16B aligned.
DEVINL v16h load2x8(const _Float16* rp) {
    union { uint4 u[2]; v16h h; } t;
    t.u[0] = *(const uint4*)(rp);
    t.u[1] = *(const uint4*)(rp + 16);
    return t.h;
}

// ---------------------------------------------------------------------------
// 1x1 conv as per-batch GEMM:  O[b,m,n] = sum_k A[m,k] * Xt[b,n,k]  (+bias)
// A: [M16,K] row-major, zero-padded to a multiple of 16 rows -> unconditional loads.
// Xt: [B][HW][K] NHWC (k contiguous).
// Each wave computes two 16x16 n-subtiles (n0, n0+64) sharing one A fragment.
// omode: 0 = padded-NHWC f16 out (66x66 halo) + NCHW f32 residual (nonlocal W-conv)
//        1 = NCHW f16 out (theta -> T matrix rows)
//        2 = transposed scatter Pt[j][i]   (phi)
//        3 = transposed scatter Gt[cg][jr] (g)
//        4 = NCHW f32 out (head 1x1 tails), rows m >= M masked
// grid: (HW/128, ceil(M/16), B), block 128 (4 waves)
// ---------------------------------------------------------------------------
__global__ __launch_bounds__(128) void conv1x1_wmma(
        const _Float16* __restrict__ A, const _Float16* __restrict__ Xt,
        const float* __restrict__ bias, const float* __restrict__ resid,
        float* __restrict__ outF, _Float16* __restrict__ outH,
        int M, int Kd, int HW, int omode)
{
    int wave = threadIdx.x >> 5;
    int lane = threadIdx.x & 31;
    int half = lane >> 4;
    int b  = blockIdx.z;
    int m0 = blockIdx.y * 16;
    int n0 = blockIdx.x * 128 + wave * 16;

    const _Float16* arp = A + (size_t)(m0 + (lane & 15)) * Kd + 8 * half;
    const _Float16* xr0 = Xt + ((size_t)b * HW + n0 + (lane & 15)) * Kd + 8 * half;
    const _Float16* xr1 = xr0 + (size_t)64 * Kd;

    v8f c0 = {}, c1 = {};
    for (int k0 = 0; k0 < Kd; k0 += 32) {
        v16h a = load2x8(arp + k0);
        c0 = wmma_f16(a, load2x8(xr0 + k0), c0);
        c1 = wmma_f16(a, load2x8(xr1 + k0), c1);
    }

    #pragma unroll
    for (int t = 0; t < 2; ++t) {
        int col = n0 + (lane & 15) + t * 64;
        v8f c = t ? c1 : c0;
        if (omode == 0) {
            union { uint4 u; _Float16 h[8]; } pk;
            #pragma unroll
            for (int r = 0; r < 8; ++r) {
                int m = m0 + 8 * half + r;
                float v = c[r] + bias[m] + resid[((size_t)b * M + m) * HW + col];
                pk.h[r] = (_Float16)v;
            }
            int yy = col >> 6, xx = col & 63;
            *(uint4*)(outH + (((size_t)b * 66 + yy + 1) * 66 + xx + 1) * M + m0 + 8 * half) = pk.u;
        } else if (omode == 1) {
            #pragma unroll
            for (int r = 0; r < 8; ++r) {
                int m = m0 + 8 * half + r;
                outH[((size_t)b * M + m) * HW + col] = (_Float16)(c[r] + bias[m]);
            }
        } else if (omode == 2) {
            #pragma unroll
            for (int r = 0; r < 8; ++r) {
                int m = m0 + 8 * half + r;
                size_t f = ((size_t)b * M + m) * HW + col;        // flat NCHW index
                outH[(f & 8191) * 128 + (f >> 13)] = (_Float16)(c[r] + bias[m]);
            }
        } else if (omode == 3) {
            #pragma unroll
            for (int r = 0; r < 8; ++r) {
                int m = m0 + 8 * half + r;
                size_t f = ((size_t)b * M + m) * HW + col;
                outH[(f & 127) * 8192 + (f >> 7)] = (_Float16)(c[r] + bias[m]);
            }
        } else {
            #pragma unroll
            for (int r = 0; r < 8; ++r) {
                int m = m0 + 8 * half + r;
                if (m < M) outF[((size_t)b * M + m) * HW + col] = c[r] + bias[m];
            }
        }
    }
}

// ---------------------------------------------------------------------------
// 3x3 conv (pad 1), zero-halo padded NHWC f16 input [B][66][66][C], implicit
// im2col, bias + relu, unpadded NHWC f16 out. All fragments are unconditional
// 2x b128 global loads. grid: (C/64, H, B), block 256 (8 waves).
// Workgroup computes 64 co x 64 px (one image row y).
// ---------------------------------------------------------------------------
__global__ __launch_bounds__(256) void conv3x3_wmma(
        const _Float16* __restrict__ Wt, const _Float16* __restrict__ Xp,
        const float* __restrict__ bias, _Float16* __restrict__ outH, int C)
{
    int tid = threadIdx.x, wave = tid >> 5, lane = tid & 31, half = lane >> 4;
    int cob = blockIdx.x, y = blockIdx.y, b = blockIdx.z;
    int m0  = cob * 64 + (wave >> 1) * 16;
    int px0 = (wave & 1) * 32;
    int xl0 = px0 + (lane & 15);

    v8f acc0 = {}, acc1 = {};
    for (int cc = 0; cc < C; cc += 32) {
        #pragma unroll
        for (int dy = 0; dy < 3; ++dy) {
            // padded row index = y + dy (in [0,65])
            const _Float16* xrow = Xp + (((size_t)b * 66 + y + dy) * 66) * C + cc + 8 * half;
            #pragma unroll
            for (int dx = 0; dx < 3; ++dx) {
                const _Float16* wrp =
                    Wt + ((size_t)(dy * 3 + dx) * C + m0 + (lane & 15)) * C + cc + 8 * half;
                v16h a = load2x8(wrp);
                // padded col index = xl0 + dx (in [0,65])
                const _Float16* xp0 = xrow + (size_t)(xl0 + dx) * C;
                v16h b0 = load2x8(xp0);
                v16h b1 = load2x8(xp0 + (size_t)16 * C);
                acc0 = wmma_f16(a, b0, acc0);
                acc1 = wmma_f16(a, b1, acc1);
            }
        }
    }
    union { uint4 u; _Float16 h[8]; } p0, p1;
    #pragma unroll
    for (int r = 0; r < 8; ++r) {
        int m = m0 + 8 * half + r;
        float bv = bias[m];
        p0.h[r] = (_Float16)fmaxf(acc0[r] + bv, 0.0f);
        p1.h[r] = (_Float16)fmaxf(acc1[r] + bv, 0.0f);
    }
    size_t o0 = ((size_t)(b * 64 + y) * 64 + xl0) * C + m0 + 8 * half;
    *(uint4*)(outH + o0)                  = p0.u;
    *(uint4*)(outH + o0 + 16 * (size_t)C) = p1.u;
}

// ---------------------------------------------------------------------------
// Flash attention over the whole-batch reshape semantics:
//   Y[8192,128] = softmax(T[8192,128] @ P[128,8192], axis=1) @ G[8192,128]
// T:  [N][128] row-major (k contiguous)        -> A frags from global
// Pt: [N][128]  Pt[j][i] = Pmat[i][j]          -> score B frags from global
// Gt: [128][N]  Gt[cg][jr] = Gmat[jr][cg]      -> O-update B frags from global
// Yt: [B*HW][128] NHWC output for the following W-conv.
// grid: N/64 blocks of 128 threads (4 waves x 16 rows). Online softmax, no barriers.
// ---------------------------------------------------------------------------
__global__ __launch_bounds__(128) void flash_attn_wmma(
        const _Float16* __restrict__ T, const _Float16* __restrict__ Pt,
        const _Float16* __restrict__ Gt, _Float16* __restrict__ Yt, int N)
{
    __shared__ _Float16 es[4 * 16 * 32];   // per-wave exp transpose patch, 4 KB
    int tid = threadIdx.x, wave = tid >> 5, lane = tid & 31, half = lane >> 4;
    int row0 = blockIdx.x * 64 + wave * 16;

    const _Float16* trow = T + (size_t)(row0 + (lane & 15)) * 128 + 8 * half;
    v16h at[4];
    #pragma unroll
    for (int kk = 0; kk < 4; ++kk) at[kk] = load2x8(trow + kk * 32);

    float mrun[8], lrun[8];
    v8f o[8];
    v8f vz = {};
    #pragma unroll
    for (int r = 0; r < 8; ++r) { mrun[r] = -1e30f; lrun[r] = 0.0f; }
    #pragma unroll
    for (int nt = 0; nt < 8; ++nt) o[nt] = vz;
    _Float16* ew = es + wave * (16 * 32);
    const _Float16* grow0 = Gt + (size_t)(lane & 15) * N + 8 * half;

    for (int j = 0; j < N; j += 32) {
        const _Float16* prow0 = Pt + (size_t)(j + (lane & 15)) * 128 + 8 * half;
        const _Float16* prow1 = prow0 + 16 * 128;
        if (j + 32 < N) {
            __builtin_prefetch(prow0 + 32 * 128, 0, 0);
            __builtin_prefetch(grow0 + j + 32, 0, 0);
        }
        v8f s0 = {}, s1 = {};
        #pragma unroll
        for (int kk = 0; kk < 4; ++kk) {
            s0 = wmma_f16(at[kk], load2x8(prow0 + kk * 32), s0);
            s1 = wmma_f16(at[kk], load2x8(prow1 + kk * 32), s1);
        }
        float scale[8];
        #pragma unroll
        for (int r = 0; r < 8; ++r) {
            float mx = fmaxf(s0[r], s1[r]);
            mx = fmaxf(mx, __shfl_xor(mx, 1, 32));
            mx = fmaxf(mx, __shfl_xor(mx, 2, 32));
            mx = fmaxf(mx, __shfl_xor(mx, 4, 32));
            mx = fmaxf(mx, __shfl_xor(mx, 8, 32));
            float mnew = fmaxf(mrun[r], mx);
            scale[r] = __expf(mrun[r] - mnew);
            mrun[r] = mnew;
            float e0 = __expf(s0[r] - mnew);
            float e1 = __expf(s1[r] - mnew);
            float sum = e0 + e1;
            sum += __shfl_xor(sum, 1, 32);
            sum += __shfl_xor(sum, 2, 32);
            sum += __shfl_xor(sum, 4, 32);
            sum += __shfl_xor(sum, 8, 32);
            lrun[r] = lrun[r] * scale[r] + sum;
            int mm = r + 8 * half;
            ew[mm * 32 + (lane & 15)]      = (_Float16)e0;
            ew[mm * 32 + (lane & 15) + 16] = (_Float16)e1;
        }
        #pragma unroll
        for (int nt = 0; nt < 8; ++nt) {
            #pragma unroll
            for (int r = 0; r < 8; ++r) o[nt][r] *= scale[r];
        }
        v16h ae = load2x8(ew + (lane & 15) * 32 + 8 * half);
        #pragma unroll
        for (int nt = 0; nt < 8; ++nt) {
            v16h bg = load2x8(grow0 + (size_t)nt * 16 * N + j);
            o[nt] = wmma_f16(ae, bg, o[nt]);
        }
    }
    float linv[8];
    #pragma unroll
    for (int r = 0; r < 8; ++r) linv[r] = 1.0f / lrun[r];
    #pragma unroll
    for (int nt = 0; nt < 8; ++nt) {
        #pragma unroll
        for (int r = 0; r < 8; ++r) {
            size_t f = (size_t)(row0 + r + 8 * half) * 128 + nt * 16 + (lane & 15);
            int bb  = (int)(f >> 19);           // / 524288
            size_t rem = f & 524287;
            int ci = (int)(rem >> 12);
            int n  = (int)(rem & 4095);
            Yt[((size_t)bb * 4096 + n) * 128 + ci] = (_Float16)(o[nt][r] * linv[r]);
        }
    }
}

// ---------------------------------------------------------------------------
// Helpers: conversion + weight permutation + fill + gather
// ---------------------------------------------------------------------------
__global__ void cvt_f16_kernel(const float* __restrict__ in, _Float16* __restrict__ out, int n) {
    int i = blockIdx.x * 256 + threadIdx.x;
    if (i < n) out[i] = (_Float16)in[i];
}

// convert with zero tail padding up to ntot
__global__ void cvt_pad_kernel(const float* __restrict__ in, _Float16* __restrict__ out,
                               int n, int ntot) {
    int i = blockIdx.x * 256 + threadIdx.x;
    if (i < ntot) out[i] = (i < n) ? (_Float16)in[i] : (_Float16)0.0f;
}

// f32 NCHW -> f16 NHWC
__global__ void cvt_nhwc_kernel(const float* __restrict__ in, _Float16* __restrict__ out,
                                int C, int HW, int B) {
    int i = blockIdx.x * 256 + threadIdx.x;
    int total = B * C * HW;
    if (i >= total) return;
    int n = i % HW;
    int c = (i / HW) % C;
    int b = i / (HW * C);
    out[((size_t)b * HW + n) * C + c] = (_Float16)in[i];
}

// conv3x3 weights (co,ci,3,3) -> [tap][co][ci]
__global__ void cvt_w3x3_kernel(const float* __restrict__ w, _Float16* __restrict__ o, int C) {
    int idx = blockIdx.x * 256 + threadIdx.x;
    int total = C * C * 9;
    if (idx >= total) return;
    int tap = idx % 9;
    int rem = idx / 9;
    int ci  = rem % C;
    int co  = rem / C;
    o[((size_t)tap * C + co) * C + ci] = (_Float16)w[idx];
}

__global__ void fill0_kernel(_Float16* __restrict__ p, int n8) {
    int i = blockIdx.x * 256 + threadIdx.x;
    if (i < n8) {
        uint4 z = {0u, 0u, 0u, 0u};
        *(uint4*)(p + (size_t)i * 8) = z;
    }
}

__global__ void gather_kernel(const float* __restrict__ src, const int* __restrict__ ind,
                              float* __restrict__ dst, int Cd, int K, int HW, int B) {
    int i = blockIdx.x * 256 + threadIdx.x;
    int total = B * K * Cd;
    if (i >= total) return;
    int c = i % Cd;
    int k = (i / Cd) % K;
    int b = i / (Cd * K);
    int hw = ind[b * K + k];
    if (hw < 0) hw = 0;
    if (hw >= HW) hw = HW - 1;
    dst[i] = src[((size_t)b * Cd + c) * HW + hw];
}

// ---------------------------------------------------------------------------
extern "C" void kernel_launch(void* const* d_in, const int* in_sizes, int n_in,
                              void* d_out, int out_size, void* d_ws, size_t ws_size,
                              hipStream_t stream) {
    (void)in_sizes; (void)n_in; (void)out_size; (void)ws_size;
    const int B = 2, C = 256, Cb = 128, H = 64, HW = 4096, K = 128, NROWS = 8192;
    const int PADHW = 66 * 66;   // zero-halo padded image

    const float* feature = (const float*)d_in[0];
    const int*   tl_tags = (const int*)d_in[1];
    const int*   br_tags = (const int*)d_in[2];

    char* ws = (char*)d_ws;
    size_t off = 0;
    auto alloc = [&](size_t bytes) -> void* {
        void* p = ws + off;
        off += (bytes + 255) & ~(size_t)255;
        return p;
    };

    _Float16* xt = (_Float16*)alloc((size_t)B * C * HW * 2);   // feature, NHWC f16
    _Float16* nlw[2][4];
    for (int i = 0; i < 2; ++i)
        for (int j = 0; j < 4; ++j) nlw[i][j] = (_Float16*)alloc((size_t)Cb * C * 2);
    _Float16* w1h[6];
    _Float16* w2h[6];
    for (int i = 0; i < 6; ++i) w1h[i] = (_Float16*)alloc((size_t)C * C * 9 * 2);
    for (int i = 0; i < 6; ++i) w2h[i] = (_Float16*)alloc((size_t)80 * C * 2);  // >= 16-row pad
    _Float16* Tb = (_Float16*)alloc((size_t)NROWS * Cb * 2);
    _Float16* Pt = (_Float16*)alloc((size_t)NROWS * Cb * 2);
    _Float16* Gt = (_Float16*)alloc((size_t)NROWS * Cb * 2);
    _Float16* Yt = (_Float16*)alloc((size_t)NROWS * Cb * 2);
    _Float16* nlh[2];
    nlh[0] = (_Float16*)alloc((size_t)B * PADHW * C * 2);      // padded NHWC
    nlh[1] = (_Float16*)alloc((size_t)B * PADHW * C * 2);      // padded NHWC
    _Float16* hb = (_Float16*)alloc((size_t)B * C * HW * 2);   // NHWC
    float* tmp32 = (float*)alloc((size_t)B * 2 * HW * 4);

    auto cvt = [&](const float* src, _Float16* dst, int n) {
        cvt_f16_kernel<<<(n + 255) / 256, 256, 0, stream>>>(src, dst, n);
    };

    // --- precision conversion / layout passes ---
    cvt_nhwc_kernel<<<(B * C * HW + 255) / 256, 256, 0, stream>>>(feature, xt, C, HW, B);
    for (int blk = 0; blk < 2; ++blk) {
        int base = 3 + blk * 8;
        cvt((const float*)d_in[base + 0], nlw[blk][0], Cb * C);  // theta_w
        cvt((const float*)d_in[base + 2], nlw[blk][1], Cb * C);  // phi_w
        cvt((const float*)d_in[base + 4], nlw[blk][2], Cb * C);  // g_w
        cvt((const float*)d_in[base + 6], nlw[blk][3], C * Cb);  // W_w
    }
    const int head_base[6] = {19, 23, 27, 31, 35, 39};
    const int head_out[6]  = {80, 80, 1, 1, 2, 2};
    for (int h = 0; h < 6; ++h) {
        cvt_w3x3_kernel<<<(C * C * 9 + 255) / 256, 256, 0, stream>>>(
            (const float*)d_in[head_base[h]], w1h[h], C);
        int mpad = ((head_out[h] + 15) / 16) * 16;
        cvt_pad_kernel<<<(mpad * C + 255) / 256, 256, 0, stream>>>(
            (const float*)d_in[head_base[h] + 2], w2h[h], head_out[h] * C, mpad * C);
    }
    // zero-fill padded nonlocal output buffers (halo stays zero; interior rewritten)
    {
        int n8 = (B * PADHW * C) / 8;
        fill0_kernel<<<(n8 + 255) / 256, 256, 0, stream>>>(nlh[0], n8);
        fill0_kernel<<<(n8 + 255) / 256, 256, 0, stream>>>(nlh[1], n8);
    }

    // --- non-local blocks (tl, br) ---
    for (int blk = 0; blk < 2; ++blk) {
        int base = 3 + blk * 8;
        dim3 g1(HW / 128, Cb / 16, B);
        conv1x1_wmma<<<g1, 128, 0, stream>>>(nlw[blk][0], xt, (const float*)d_in[base + 1],
                                             nullptr, nullptr, Tb, Cb, C, HW, 1);
        conv1x1_wmma<<<g1, 128, 0, stream>>>(nlw[blk][1], xt, (const float*)d_in[base + 3],
                                             nullptr, nullptr, Pt, Cb, C, HW, 2);
        conv1x1_wmma<<<g1, 128, 0, stream>>>(nlw[blk][2], xt, (const float*)d_in[base + 5],
                                             nullptr, nullptr, Gt, Cb, C, HW, 3);
        flash_attn_wmma<<<dim3(NROWS / 64), 128, 0, stream>>>(Tb, Pt, Gt, Yt, NROWS);
        dim3 g2(HW / 128, C / 16, B);
        conv1x1_wmma<<<g2, 128, 0, stream>>>(nlw[blk][3], Yt, (const float*)d_in[base + 7],
                                             feature, nullptr, nlh[blk], C, Cb, HW, 0);
    }

    // --- heads ---
    float* outp    = (float*)d_out;
    float* heat0   = outp;
    float* heat1   = outp + 655360;
    float* tagout0 = outp + 1310720;
    float* tagout1 = outp + 1310976;
    float* regr0   = outp + 1311232;
    float* regr1   = outp + 1311744;

    for (int h = 0; h < 6; ++h) {
        const _Float16* src = nlh[h & 1];   // tl/br alternate
        conv3x3_wmma<<<dim3(C / 64, H, B), 256, 0, stream>>>(
            w1h[h], src, (const float*)d_in[head_base[h] + 1], hb, C);
        int M = head_out[h];
        dim3 g(HW / 128, (M + 15) / 16, B);
        if (h < 2) {
            float* dst = (h == 0) ? heat0 : heat1;
            conv1x1_wmma<<<g, 128, 0, stream>>>(w2h[h], hb, (const float*)d_in[head_base[h] + 3],
                                                nullptr, dst, nullptr, M, C, HW, 4);
        } else {
            conv1x1_wmma<<<g, 128, 0, stream>>>(w2h[h], hb, (const float*)d_in[head_base[h] + 3],
                                                nullptr, tmp32, nullptr, M, C, HW, 4);
            const int* ind = (h & 1) ? br_tags : tl_tags;
            float* dst = (h == 2) ? tagout0 : (h == 3) ? tagout1 : (h == 4) ? regr0 : regr1;
            int total = B * K * M;
            gather_kernel<<<(total + 255) / 256, 256, 0, stream>>>(tmp32, ind, dst, M, K, HW, B);
        }
    }
}
